// CrissCrossAtten_20976620273854
// MI455X (gfx1250) — matrix operand order repaired and checked by reading, not compiled, not run on hardware
//
#include <hip/hip_runtime.h>

// ---------------------------------------------------------------------------
// CrissCross attention, fused, CDNA5 (gfx1250) WMMA bf16 path.
//   scores = Q K^T            (kept in LDS as bf16, full 32 x 2048 strip)
//   mixed  = scores @ map     (dominant GEMM; map pre-transposed to bf16)
//   attn   = softmax(mask(0.125 * mixed))   (online/flash softmax per wave)
//   out    = attn @ V         (V pre-transposed to bf16 [b,h,d,s])
// Each wave processes BOTH 16-row blocks and a 64-wide t-chunk per step:
// per ss iteration 8 WMMAs are fed by 4 LDS-b128 + 8 global-b128 loads.
// ---------------------------------------------------------------------------

typedef __attribute__((ext_vector_type(16))) __bf16 v16bf;
typedef __attribute__((ext_vector_type(8)))  float  v8f;

#define BB 4
#define LL 2048
#define HH 8
#define EE 64
#define DD 64
#define SN 2048            // S dimension
#define SSTRIDE 2056       // LDS score row stride (mult of 8 -> 16B aligned groups)
#define NEG_BIG (-3.0e38f) // finite "-inf" so online softmax never produces NaN

// B-operand / contiguous-K load: elems 0..15 <- p[0..15] (lane-half offset applied by caller)
static __device__ inline v16bf ld16_bf(const __bf16* p) {
  union { uint4 u[2]; v16bf v; } c;
  c.u[0] = *(const uint4*)(p);
  c.u[1] = *(const uint4*)(p + 8);
  return c.v;
}
// A-operand load (bf16 source): elems 0..15 <- p[s0..s0+15]
static __device__ inline v16bf ldA_bf(const __bf16* p, int s0) {
  union { uint4 u[2]; v16bf v; } c;
  c.u[0] = *(const uint4*)(p + s0);
  c.u[1] = *(const uint4*)(p + s0 + 8);
  return c.v;
}
// A-operand from f32 global (contiguous 16 K values, convert to bf16)
static __device__ inline v16bf ldA_f32(const float* p, int e0) {
  v16bf a;
#pragma unroll
  for (int i = 0; i < 8; ++i) a[i]     = (__bf16)p[e0 + i];
#pragma unroll
  for (int i = 0; i < 8; ++i) a[i + 8] = (__bf16)p[e0 + 8 + i];
  return a;
}
// B-operand from f32 global (contiguous 16 K values, convert)
static __device__ inline v16bf ldB_f32(const float* p, int e0) {
  v16bf b;
#pragma unroll
  for (int i = 0; i < 16; ++i) b[i] = (__bf16)p[e0 + i];
  return b;
}
static __device__ inline v8f wmma_bf16(v16bf a, v16bf b, v8f c) {
  return __builtin_amdgcn_wmma_f32_16x16x32_bf16(false, a, false, b, (short)0, c,
                                                 false, false);
}

// --------------------------- prep kernels ----------------------------------
// mapT[t*2048 + s] = bf16(map[s*2048 + t])
__global__ void prep_map_kernel(const float* __restrict__ mp, __bf16* __restrict__ mapT) {
  int idx = blockIdx.x * 256 + threadIdx.x;
  int t = idx >> 11, s = idx & 2047;
  mapT[idx] = (__bf16)mp[s * SN + t];
}
// Vt[((b*H+h)*D + d)*S + s] = bf16(V[((b*S+s)*H+h)*D + d])
__global__ void prep_v_kernel(const float* __restrict__ V, __bf16* __restrict__ Vt) {
  int idx = blockIdx.x * 256 + threadIdx.x;
  int s = idx & 2047;
  int d = (idx >> 11) & 63;
  int h = (idx >> 17) & 7;
  int b = idx >> 20;
  Vt[idx] = (__bf16)V[(((size_t)b * SN + s) * HH + h) * DD + d];
}

// --------------------------- main fused kernel -----------------------------
__global__ __launch_bounds__(256) void cc_attn_kernel(
    const float* __restrict__ Q, const float* __restrict__ K,
    const __bf16* __restrict__ mapT, const __bf16* __restrict__ Vt,
    float* __restrict__ out) {
  __shared__ __bf16 sc[32 * SSTRIDE];       // 128.5 KB score strip (bf16)
  __shared__ __bf16 pst[8][32][64];         // per-wave P relayout bounce (32 KB)
  __shared__ float  mO[8][32][64];          // per-wave partial O (64 KB)
  __shared__ float  mMax[8][32];
  __shared__ float  mSum[8][32];

  const int tid  = threadIdx.x;
  const int lane = tid & 31;
  const int w    = tid >> 5;       // wave id 0..7
  const int l16  = lane & 15;
  const int hiK  = lane >> 4;      // lane-half (K-split)

  const int tilesPerHead = LL / 32;                          // 64
  const int tI = blockIdx.x % tilesPerHead;
  const int hh = (blockIdx.x / tilesPerHead) % HH;
  const int bb = blockIdx.x / (tilesPerHead * HH);
  const int l0 = (tilesPerHead - 1 - tI) * 32;               // longest tiles first

  // ---------------- Phase 1: scores strip  sc[32 x 2048] = Q K^T ----------
  {
    const float* qbase = Q + (((size_t)bb * LL + l0) * HH + hh) * EE;
#pragma unroll 1
    for (int it = 0; it < 32; ++it) {
      int tt = w + it * 8;           // 0..255 tiles (2 rowblocks x 128 coltiles)
      int rbp = tt >> 7;
      int n0 = (tt & 127) * 16;
      const float* qrow = qbase + (size_t)(rbp * 16 + l16) * HH * EE;
      const float* krow = K + (((size_t)bb * LL + (n0 + l16)) * HH + hh) * EE;
      v8f acc;
#pragma unroll
      for (int r = 0; r < 8; ++r) acc[r] = 0.0f;
#pragma unroll
      for (int ks = 0; ks < 2; ++ks) {
        v16bf a = ldA_f32(qrow, ks * 32 + hiK * 16);
        v16bf b = ldB_f32(krow, ks * 32 + hiK * 16);
        acc = wmma_bf16(a, b, acc);
      }
#pragma unroll
      for (int r = 0; r < 8; ++r)
        sc[(rbp * 16 + r + hiK * 8) * SSTRIDE + n0 + l16] = (__bf16)acc[r];
    }
  }
  __syncthreads();

  // ------- Phase 2+3: mixed GEMM + online softmax + P.V (both rowblocks) --
  const int nchunks = l0 / 64 + 1;     // 64-wide t-chunks under causal mask

  float rmax[2][8], rsum[2][8];
  v8f o[2][4];
#pragma unroll
  for (int rb2 = 0; rb2 < 2; ++rb2) {
#pragma unroll
    for (int r = 0; r < 8; ++r) { rmax[rb2][r] = -1.0e30f; rsum[rb2][r] = 0.0f; }
#pragma unroll
    for (int dt = 0; dt < 4; ++dt)
#pragma unroll
      for (int r = 0; r < 8; ++r) o[rb2][dt][r] = 0.0f;
  }

  const __bf16* scrow0 = sc + (size_t)l16 * SSTRIDE;
  const __bf16* scrow1 = sc + (size_t)(16 + l16) * SSTRIDE;
  const __bf16* vbase  = Vt + (((size_t)bb * HH + hh) * DD) * (size_t)SN;

  for (int j = w; j < nchunks; j += 8) {
    const int t0 = j * 64;
    v8f m[2][4];
#pragma unroll
    for (int rb2 = 0; rb2 < 2; ++rb2)
#pragma unroll
      for (int k = 0; k < 4; ++k)
#pragma unroll
        for (int r = 0; r < 8; ++r) m[rb2][k][r] = 0.0f;

    const __bf16* bt[4];
#pragma unroll
    for (int k = 0; k < 4; ++k)
      bt[k] = mapT + (size_t)(t0 + k * 16 + l16) * SN + hiK * 16;

    // warm L2/WGP path for this wave's next chunk (CDNA5 global_prefetch_b8)
    if (j + 8 < nchunks) {
      __builtin_prefetch(bt[0] + (size_t)512 * SN, 0, 1);
      __builtin_prefetch(bt[2] + (size_t)512 * SN, 0, 1);
    }

#pragma unroll 2
    for (int ss = 0; ss < 64; ++ss) {
      const int s0 = ss * 32;
      v16bf a0 = ldA_bf(scrow0, s0 + hiK * 16);
      v16bf a1 = ldA_bf(scrow1, s0 + hiK * 16);
#pragma unroll
      for (int k = 0; k < 4; ++k) {
        v16bf bk = ld16_bf(bt[k] + s0);
        m[0][k] = wmma_bf16(a0, bk, m[0][k]);
        m[1][k] = wmma_bf16(a1, bk, m[1][k]);
      }
    }

    // online softmax update per rowblock (64 t-columns this chunk)
#pragma unroll
    for (int rb2 = 0; rb2 < 2; ++rb2) {
#pragma unroll
      for (int r = 0; r < 8; ++r) {
        const int lrow = l0 + rb2 * 16 + r + hiK * 8;
        float v[4];
#pragma unroll
        for (int k = 0; k < 4; ++k)
          v[k] = ((t0 + k * 16 + l16) > lrow) ? NEG_BIG : m[rb2][k][r] * 0.125f;
        float bm = fmaxf(fmaxf(v[0], v[1]), fmaxf(v[2], v[3]));
        bm = fmaxf(bm, __shfl_xor(bm, 1, 32));
        bm = fmaxf(bm, __shfl_xor(bm, 2, 32));
        bm = fmaxf(bm, __shfl_xor(bm, 4, 32));
        bm = fmaxf(bm, __shfl_xor(bm, 8, 32));
        const float mnew = fmaxf(rmax[rb2][r], bm);
        const float fac  = __expf(rmax[rb2][r] - mnew);
        float p[4];
        float rs = 0.0f;
#pragma unroll
        for (int k = 0; k < 4; ++k) { p[k] = __expf(v[k] - mnew); rs += p[k]; }
        rs += __shfl_xor(rs, 1, 32);
        rs += __shfl_xor(rs, 2, 32);
        rs += __shfl_xor(rs, 4, 32);
        rs += __shfl_xor(rs, 8, 32);
        rsum[rb2][r] = rsum[rb2][r] * fac + rs;
        rmax[rb2][r] = mnew;
        o[rb2][0][r] *= fac; o[rb2][1][r] *= fac;
        o[rb2][2][r] *= fac; o[rb2][3][r] *= fac;
#pragma unroll
        for (int k = 0; k < 4; ++k)
          pst[w][rb2 * 16 + r + hiK * 8][k * 16 + l16] = (__bf16)p[k];
      }
    }
    // LDS bounce C-layout -> A-layout; DS in-order per wave, wait for stores
    asm volatile("s_wait_dscnt 0x0" ::: "memory");

    // P.V accumulate: K = 64 (two 32-wide k-steps), V tiles shared by rowblocks
#pragma unroll
    for (int ks = 0; ks < 2; ++ks) {
      v16bf pa0 = ldA_bf(&pst[w][l16][0],      ks * 32 + hiK * 16);
      v16bf pa1 = ldA_bf(&pst[w][16 + l16][0], ks * 32 + hiK * 16);
#pragma unroll
      for (int dt = 0; dt < 4; ++dt) {
        const __bf16* vrow =
            vbase + (size_t)(dt * 16 + l16) * SN + t0 + ks * 32 + hiK * 16;
        v16bf bv = ld16_bf(vrow);
        o[0][dt] = wmma_bf16(pa0, bv, o[0][dt]);
        o[1][dt] = wmma_bf16(pa1, bv, o[1][dt]);
      }
    }
  }

  // ---------------- dump partials, merge 8 waves per row ------------------
#pragma unroll
  for (int rb2 = 0; rb2 < 2; ++rb2) {
#pragma unroll
    for (int dt = 0; dt < 4; ++dt)
#pragma unroll
      for (int r = 0; r < 8; ++r)
        mO[w][rb2 * 16 + r + hiK * 8][dt * 16 + l16] = o[rb2][dt][r];
    if (l16 == 0) {
#pragma unroll
      for (int r = 0; r < 8; ++r) {
        mMax[w][rb2 * 16 + r + hiK * 8] = rmax[rb2][r];
        mSum[w][rb2 * 16 + r + hiK * 8] = rsum[rb2][r];
      }
    }
  }
  __syncthreads();

  for (int p = tid; p < 32 * 64; p += 256) {
    const int row = p >> 6;
    const int d   = p & 63;
    float M = NEG_BIG;
#pragma unroll
    for (int w2 = 0; w2 < 8; ++w2) M = fmaxf(M, mMax[w2][row]);
    float Z = 0.0f, num = 0.0f;
#pragma unroll
    for (int w2 = 0; w2 < 8; ++w2) {
      const float e = __expf(mMax[w2][row] - M);
      Z   += e * mSum[w2][row];
      num += e * mO[w2][row][d];
    }
    out[(((size_t)bb * LL + (l0 + row)) * HH + hh) * DD + d] = num / fmaxf(Z, 1e-37f);
  }
}

// ---------------------------------------------------------------------------
extern "C" void kernel_launch(void* const* d_in, const int* in_sizes, int n_in,
                              void* d_out, int out_size, void* d_ws, size_t ws_size,
                              hipStream_t stream) {
  const float* Q  = (const float*)d_in[0];
  const float* Kp = (const float*)d_in[1];
  const float* V  = (const float*)d_in[2];
  const float* mp = (const float*)d_in[3];
  float* out = (float*)d_out;

  __bf16* mapT = (__bf16*)d_ws;                                    // 8 MB
  __bf16* Vt   = (__bf16*)((char*)d_ws + (size_t)8 * 1024 * 1024); // 8 MB

  const int mapElems = SN * SN;                 // 4M
  const int vElems   = BB * HH * DD * SN;       // 4M
  prep_map_kernel<<<mapElems / 256, 256, 0, stream>>>(mp, mapT);
  prep_v_kernel<<<vElems / 256, 256, 0, stream>>>(V, Vt);

  const int grid = BB * HH * (LL / 32);         // 2048 workgroups
  cc_attn_kernel<<<grid, 256, 0, stream>>>(Q, Kp, mapT, Vt, out);
}
